// Mamba_2379411882214
// MI455X (gfx1250) — matrix operand order repaired and checked by reading, not compiled, not run on hardware
//
#include <hip/hip_runtime.h>
#include <hip/hip_bf16.h>
#include <cstdint>
#include <cstddef>

// ---------------- Types for WMMA fragments (wave32, gfx1250) ----------------
typedef __attribute__((ext_vector_type(16))) __bf16          bf16x16;
typedef __attribute__((ext_vector_type(8)))  float           f32x8;
typedef __attribute__((ext_vector_type(8)))  unsigned short  u16x8;

union Frag {
    struct { u16x8 lo, hi; } h;
    bf16x16 v;
};

// ---------------- Problem constants ----------------
constexpr int B_   = 4;
constexpr int L_   = 2048;
constexpr int DM   = 1024;   // d_model
constexpr int DI   = 2048;   // d_inner
constexpr int DS   = 16;     // d_state
constexpr int DR   = 64;     // dt_rank
constexpr int T_   = B_ * L_;       // 8192 tokens
constexpr int NXZ  = 2 * DI;        // 4096
constexpr int XDBL_LD = 128;        // padded (dt_rank 64 | B 16 | C 16 | pad 32)

// ---------------- helpers ----------------
__device__ __forceinline__ unsigned short f2bf(float f) {
    unsigned int u = __builtin_bit_cast(unsigned int, f);
    u += 0x7fffu + ((u >> 16) & 1u);           // round-to-nearest-even
    return (unsigned short)(u >> 16);
}

// A fragment: 16x32 bf16, M rows, K contiguous in memory (ld = K)
// lane (g = lane>>4, mr = lane&15): row = row0+mr, chunks [8g,8g+8) and [16+8g,+8)
__device__ __forceinline__ void load_a(Frag& f, const unsigned short* base,
                                       int ld, int row, int k, int g) {
    const unsigned short* p = base + (size_t)row * ld + k + 8 * g;
    f.h.lo = *reinterpret_cast<const u16x8*>(p);
    f.h.hi = *reinterpret_cast<const u16x8*>(p + 16);
}

// B fragment: 32x16 bf16 (KxN), stored as W[n,k] k-contiguous (ld = K)
// lane (g, nr): column n = row0+nr, chunks [8g,8g+8) and [16+8g,+8)
__device__ __forceinline__ void load_b(Frag& f, const unsigned short* base,
                                       int ld, int row, int k, int g) {
    const unsigned short* p = base + (size_t)row * ld + k + 8 * g;
    f.h.lo = *reinterpret_cast<const u16x8*>(p);
    f.h.hi = *reinterpret_cast<const u16x8*>(p + 16);
}

// ---------------- f32 -> bf16 conversion with zero tail-padding ----------------
__global__ __launch_bounds__(256)
void mamba_f32_to_bf16_pad(const float* __restrict__ src,
                           unsigned short* __restrict__ dst,
                           long long n_src, long long n_dst) {
    long long i = (long long)blockIdx.x * 256 + threadIdx.x;
    if (i < n_dst) dst[i] = (i < n_src) ? f2bf(src[i]) : (unsigned short)0;
}

// ---------------- bf16 WMMA GEMM: C[M,N] = A[M,K] * B[N,K]^T ----------------
// MODE 0: plain f32 store   MODE 2: softplus(acc + bias[n]) (for dt_proj)
template<int MODE>
__global__ __launch_bounds__(256)
void mamba_gemm_bf16(const unsigned short* __restrict__ A,
                     const unsigned short* __restrict__ B,
                     float* __restrict__ C,
                     const float* __restrict__ bias,
                     int K, int ldc) {
    const int tid  = threadIdx.x;
    const int lane = tid & 31;
    const int wid  = tid >> 5;
    const int wm   = wid & 1;          // 2 waves along M
    const int wn   = wid >> 1;         // 4 waves along N
    const int m_wave = blockIdx.y * 128 + wm * 64;   // 4 M-tiles per wave
    const int n_wave = blockIdx.x * 128 + wn * 32;   // 2 N-tiles per wave
    const int g  = lane >> 4;
    const int mr = lane & 15;

    f32x8 acc[4][2] = {};

    for (int k0 = 0; k0 < K; k0 += 32) {
        Frag af[4], bfr[2];
#pragma unroll
        for (int ti = 0; ti < 4; ++ti)
            load_a(af[ti], A, K, m_wave + 16 * ti + mr, k0, g);
#pragma unroll
        for (int tj = 0; tj < 2; ++tj)
            load_b(bfr[tj], B, K, n_wave + 16 * tj + mr, k0, g);
#pragma unroll
        for (int ti = 0; ti < 4; ++ti)
#pragma unroll
            for (int tj = 0; tj < 2; ++tj)
                acc[ti][tj] = __builtin_amdgcn_wmma_f32_16x16x32_bf16(
                    false, af[ti].v, false, bfr[tj].v,
                    (short)0, acc[ti][tj], false, false);
    }

#pragma unroll
    for (int ti = 0; ti < 4; ++ti) {
#pragma unroll
        for (int tj = 0; tj < 2; ++tj) {
            const int col = n_wave + 16 * tj + mr;
            float bv = 0.f;
            if (MODE == 2) bv = bias[col];
#pragma unroll
            for (int v = 0; v < 8; ++v) {
                const int row = m_wave + 16 * ti + 8 * g + v;
                float val = acc[ti][tj][v];
                if (MODE == 2) {
                    val += bv;
                    val = (val > 20.f) ? val : log1pf(__expf(val));  // softplus
                }
                C[(size_t)row * ldc + col] = val;
            }
        }
    }
}

// ---------------- causal depthwise conv1d (K=4) + SiLU ----------------
// reads x part of xz (ld NXZ), writes xs (f32) and xs_bf16
__global__ __launch_bounds__(256)
void mamba_conv_silu(const float* __restrict__ xz,
                     const float* __restrict__ cw,
                     const float* __restrict__ cb,
                     float* __restrict__ xs,
                     unsigned short* __restrict__ xsbf) {
    const int idx = blockIdx.x * 256 + threadIdx.x;    // over T_*DI
    const int c = idx & (DI - 1);
    const int t = idx >> 11;                           // token
    const int l = t & (L_ - 1);
    const int b = t >> 11;
    const float* xcol = xz + (size_t)(b * L_) * NXZ + c;
    float acc = cb[c];
#pragma unroll
    for (int j = 0; j < 4; ++j) {
        const int ll = l - 3 + j;
        if (ll >= 0) acc += cw[c * 4 + j] * xcol[(size_t)ll * NXZ];
    }
    const float s = acc / (1.f + __expf(-acc));        // SiLU
    xs[idx]   = s;
    xsbf[idx] = f2bf(s);
}

// ---------------- extract dt_low (cols 0..63 of x_dbl) to bf16 ----------------
__global__ __launch_bounds__(256)
void mamba_dtlow(const float* __restrict__ xdbl, unsigned short* __restrict__ dtlow) {
    const int i = blockIdx.x * 256 + threadIdx.x;      // over T_*DR
    const int r = i >> 6, col = i & 63;
    dtlow[i] = f2bf(xdbl[(size_t)r * XDBL_LD + col]);
}

// ---------------- selective scan + D-skip + SiLU(z) gating, bf16 out ----------
// thread = (channel, state): block of 256 = 16 channels x 16 states
__global__ __launch_bounds__(256)
void mamba_scan(const float* __restrict__ xs,    // T_ x DI
                const float* __restrict__ dt,    // T_ x DI
                const float* __restrict__ xdbl,  // T_ x 128 (B at 64, C at 80)
                const float* __restrict__ xz,    // T_ x NXZ (z at +DI)
                const float* __restrict__ A_log, // DI x DS
                const float* __restrict__ Dp,    // DI
                unsigned short* __restrict__ ybf) {
    const int tid = threadIdx.x;
    const int cl  = tid >> 4;
    const int n   = tid & 15;
    const int blk = blockIdx.x;                        // 0..511
    const int b   = blk >> 7;
    const int c   = ((blk & 127) << 4) + cl;

    const float Ac = -__expf(A_log[c * DS + n]);
    const float Dc = Dp[c];
    float state = 0.f;

    for (int l = 0; l < L_; ++l) {
        const int t = b * L_ + l;
        const size_t rc = (size_t)t * DI + c;
        const float dtv = dt[rc];
        const float xv  = xs[rc];
        const float Bv  = xdbl[(size_t)t * XDBL_LD + DR + n];
        const float Cv  = xdbl[(size_t)t * XDBL_LD + DR + DS + n];
        const float dA  = __expf(dtv * Ac);
        state = state * dA + (dtv * xv) * Bv;
        float y = state * Cv;
        y += __shfl_xor(y, 8, 16);
        y += __shfl_xor(y, 4, 16);
        y += __shfl_xor(y, 2, 16);
        y += __shfl_xor(y, 1, 16);
        if (n == 0) {
            const float zv = xz[(size_t)t * NXZ + DI + c];
            const float sg = zv / (1.f + __expf(-zv));
            ybf[rc] = f2bf((y + Dc * xv) * sg);
            const int lp = (l + 8 < L_) ? l + 8 : (L_ - 1);
            const size_t rp = (size_t)(b * L_ + lp) * DI + c;
            __builtin_prefetch(&dt[rp], 0, 1);          // global_prefetch_b8
            __builtin_prefetch(&xs[rp], 0, 1);
        }
    }
}

// ---------------- workspace layout (bytes) ----------------
constexpr size_t OFF_HBF    = 0;                                    // T_*DM bf16
constexpr size_t OFF_INW    = OFF_HBF    + (size_t)T_ * DM * 2;     // NXZ*DM bf16
constexpr size_t OFF_XPW    = OFF_INW    + (size_t)NXZ * DM * 2;    // 128*DI bf16 (padded)
constexpr size_t OFF_DPW    = OFF_XPW    + (size_t)128 * DI * 2;    // DI*DR bf16
constexpr size_t OFF_OPW    = OFF_DPW    + (size_t)DI * DR * 2;     // DM*DI bf16
constexpr size_t OFF_XZ     = OFF_OPW    + (size_t)DM * DI * 2;     // T_*NXZ f32
constexpr size_t OFF_XS     = OFF_XZ     + (size_t)T_ * NXZ * 4;    // T_*DI f32
constexpr size_t OFF_XSBF   = OFF_XS     + (size_t)T_ * DI * 4;     // T_*DI bf16
constexpr size_t OFF_XDBL   = OFF_XSBF   + (size_t)T_ * DI * 2;     // T_*128 f32
constexpr size_t OFF_DTLOW  = OFF_XDBL   + (size_t)T_ * XDBL_LD * 4;// T_*DR bf16
constexpr size_t OFF_DT     = OFF_DTLOW  + (size_t)T_ * DR * 2;     // T_*DI f32
constexpr size_t OFF_YBF    = OFF_DT     + (size_t)T_ * DI * 4;     // T_*DI bf16

extern "C" void kernel_launch(void* const* d_in, const int* in_sizes, int n_in,
                              void* d_out, int out_size, void* d_ws, size_t ws_size,
                              hipStream_t stream) {
    const float* h    = (const float*)d_in[0];   // hidden_states (T_, DM)
    const float* inw  = (const float*)d_in[1];   // in_proj_w (NXZ, DM)
    const float* cw   = (const float*)d_in[2];   // conv_w (DI,1,4)
    const float* cb   = (const float*)d_in[3];   // conv_b (DI)
    const float* xpw  = (const float*)d_in[4];   // x_proj_w (96, DI)
    const float* dpw  = (const float*)d_in[5];   // dt_proj_w (DI, DR)
    const float* dpb  = (const float*)d_in[6];   // dt_proj_b (DI)
    const float* alog = (const float*)d_in[7];   // A_log (DI, DS)
    const float* Dp   = (const float*)d_in[8];   // D (DI)
    const float* opw  = (const float*)d_in[9];   // out_proj_w (DM, DI)
    float* out = (float*)d_out;
    char* ws = (char*)d_ws;

    unsigned short* h_bf   = (unsigned short*)(ws + OFF_HBF);
    unsigned short* inw_bf = (unsigned short*)(ws + OFF_INW);
    unsigned short* xpw_bf = (unsigned short*)(ws + OFF_XPW);
    unsigned short* dpw_bf = (unsigned short*)(ws + OFF_DPW);
    unsigned short* opw_bf = (unsigned short*)(ws + OFF_OPW);
    float*          xz     = (float*)(ws + OFF_XZ);
    float*          xs     = (float*)(ws + OFF_XS);
    unsigned short* xs_bf  = (unsigned short*)(ws + OFF_XSBF);
    float*          xdbl   = (float*)(ws + OFF_XDBL);
    unsigned short* dtl_bf = (unsigned short*)(ws + OFF_DTLOW);
    float*          dtb    = (float*)(ws + OFF_DT);
    unsigned short* y_bf   = (unsigned short*)(ws + OFF_YBF);

    auto cvt = [&](const float* s, unsigned short* d, long long ns, long long nd) {
        int blocks = (int)((nd + 255) / 256);
        mamba_f32_to_bf16_pad<<<blocks, 256, 0, stream>>>(s, d, ns, nd);
    };

    // 1) bf16 conversions (x_proj_w zero-padded 96 -> 128 rows)
    cvt(h,   h_bf,   (long long)T_ * DM,  (long long)T_ * DM);
    cvt(inw, inw_bf, (long long)NXZ * DM, (long long)NXZ * DM);
    cvt(xpw, xpw_bf, (long long)(DR + 2 * DS) * DI, (long long)128 * DI);
    cvt(dpw, dpw_bf, (long long)DI * DR,  (long long)DI * DR);
    cvt(opw, opw_bf, (long long)DM * DI,  (long long)DM * DI);

    // 2) in_proj: xz[T_, 4096] = h * in_proj_w^T
    mamba_gemm_bf16<0><<<dim3(NXZ / 128, T_ / 128), 256, 0, stream>>>(
        h_bf, inw_bf, xz, nullptr, DM, NXZ);

    // 3) causal depthwise conv + SiLU -> xs (f32) and xs_bf16
    mamba_conv_silu<<<(T_ * DI) / 256, 256, 0, stream>>>(xz, cw, cb, xs, xs_bf);

    // 4) x_proj: xdbl[T_, 128] = xs * x_proj_w^T (padded N)
    mamba_gemm_bf16<0><<<dim3(1, T_ / 128), 256, 0, stream>>>(
        xs_bf, xpw_bf, xdbl, nullptr, DI, XDBL_LD);

    // 5) extract dt_low to bf16
    mamba_dtlow<<<(T_ * DR) / 256, 256, 0, stream>>>(xdbl, dtl_bf);

    // 6) dt_proj + softplus(.. + bias): dt[T_, DI]
    mamba_gemm_bf16<2><<<dim3(DI / 128, T_ / 128), 256, 0, stream>>>(
        dtl_bf, dpw_bf, dtb, dpb, DR, DI);

    // 7) selective scan + D-skip + SiLU(z) gating -> y_bf16
    mamba_scan<<<(B_ * DI) / 16, 256, 0, stream>>>(xs, dtb, xdbl, xz, alog, Dp, y_bf);

    // 8) out_proj: out[T_, DM] = y * out_proj_w^T
    mamba_gemm_bf16<0><<<dim3(DM / 128, T_ / 128), 256, 0, stream>>>(
        y_bf, opw_bf, out, nullptr, DI, DM);
}
